// PointNet2_MSG_50732153700751
// MI455X (gfx1250) — compile-verified, matrix operand order checked by reading, not run on hardware
//
#include <hip/hip_runtime.h>
#include <hip/hip_bf16.h>
#include <cstddef>

// ---------------------------------------------------------------------------
// PointNet++ MSG forward for MI455X (gfx1250).
// All MLP layers run through a bf16-WMMA GEMM (v_wmma_f32_16x16x32_bf16) whose
// operands are pre-packed into the CDNA5 fragment layout; each wave computes a
// 2x2 block of 16x16 tiles so the inner loop is 8x b128 loads + 4x WMMA.
// Activations use ceil32-padded channel strides with zero-filled pads => no
// K guards, no M guards (all M are multiples of 32).
// ---------------------------------------------------------------------------

#define NB 8          // batch
#define NPTS 8192     // points

typedef __attribute__((ext_vector_type(16))) __bf16 v16bf;
typedef __attribute__((ext_vector_type(8)))  float  v8f;
typedef __attribute__((ext_vector_type(8)))  unsigned int v8u;

static inline int p32(int x) { return (x + 31) & ~31; }

// pack two fp32 -> {bf16(hi),bf16(lo)} with round-to-nearest-even
static __device__ __forceinline__ unsigned int pack_bf16(float lo, float hi) {
  unsigned int a = __builtin_bit_cast(unsigned int, lo);
  unsigned int b = __builtin_bit_cast(unsigned int, hi);
  a += 0x7FFFu + ((a >> 16) & 1u);
  b += 0x7FFFu + ((b >> 16) & 1u);
  return __builtin_amdgcn_perm(b, a, 0x07060302u);  // {b[31:16], a[31:16]}
}

// ---------------------------------------------------------------------------
// pack_a: A[M, lda] fp32 (lda % 32 == 0, pads zeroed) -> fragment-ordered bf16
// Ap[((tm*nkb + kbi)*32 + lane)*8 + j]  (one uint = K pair)
// A-fragment (ISA 7.12.2, 16-bit A 16x32): lane holds row lane&15;
// lane half hf: j<4 -> K = 2j + 8*hf, j>=4 -> K = 16 + 2(j-4) + 8*hf.
// => lane needs two contiguous 8-float runs: [32*kbi+8*hf, +8) and (+16).
// ---------------------------------------------------------------------------
__global__ void pack_a(const float* __restrict__ A, unsigned int* __restrict__ Ap,
                       int lda, int nkb, int tilesM) {
  size_t total = (size_t)tilesM * nkb * 32;
  for (size_t t = blockIdx.x * (size_t)blockDim.x + threadIdx.x; t < total;
       t += (size_t)gridDim.x * blockDim.x) {
    int lane = (int)(t % 32);
    int kbi  = (int)((t / 32) % nkb);
    int tm   = (int)(t / ((size_t)32 * nkb));
    int hf   = lane >> 4;
    int row  = tm * 16 + (lane & 15);
    const float4* src = (const float4*)(A + (size_t)row * lda + (size_t)kbi * 32);
    float4 a0 = src[hf * 2 + 0];
    float4 a1 = src[hf * 2 + 1];
    float4 a2 = src[hf * 2 + 4];
    float4 a3 = src[hf * 2 + 5];
    uint4 u0, u1;
    u0.x = pack_bf16(a0.x, a0.y); u0.y = pack_bf16(a0.z, a0.w);
    u0.z = pack_bf16(a1.x, a1.y); u0.w = pack_bf16(a1.z, a1.w);
    u1.x = pack_bf16(a2.x, a2.y); u1.y = pack_bf16(a2.z, a2.w);
    u1.z = pack_bf16(a3.x, a3.y); u1.w = pack_bf16(a3.z, a3.w);
    uint4* dst = (uint4*)(Ap + t * 8);
    dst[0] = u0; dst[1] = u1;
  }
}

// pack_w: W[Kreal, Nc] fp32 -> B-fragment bf16, zero-filled to nkb*32 x ntn*16
__global__ void pack_w(const float* __restrict__ W, unsigned int* __restrict__ Wp,
                       int Kreal, int Nc, int nkb, int ntn) {
  size_t total = (size_t)ntn * nkb * 256;
  for (size_t t = blockIdx.x * (size_t)blockDim.x + threadIdx.x; t < total;
       t += (size_t)gridDim.x * blockDim.x) {
    int j    = (int)(t % 8);
    int lane = (int)((t / 8) % 32);
    int kbi  = (int)((t / 256) % nkb);
    int tn   = (int)(t / ((size_t)256 * nkb));
    int hf   = lane >> 4;
    int col  = tn * 16 + (lane & 15);
    int k0   = ((j < 4) ? (2 * j) : (16 + 2 * (j - 4))) + hf * 8 + kbi * 32;
    float w0 = 0.f, w1 = 0.f;
    if (col < Nc) {
      if (k0 < Kreal)     w0 = W[(size_t)k0 * Nc + col];
      if (k0 + 1 < Kreal) w1 = W[(size_t)(k0 + 1) * Nc + col];
    }
    Wp[t] = pack_bf16(w0, w1);
  }
}

// ---------------------------------------------------------------------------
// WMMA GEMM on pre-packed operands. 8 waves/block; each wave computes a 2x2
// block of 16x16 tiles (two M-tiles x two N-tiles). Inner loop per K-chunk:
// 8x b128 loads + 4x v_wmma_f32_16x16x32_bf16 (2 loads per WMMA).
// Stores zero-fill pad columns [Nc, ldy).
// ---------------------------------------------------------------------------
__global__ __launch_bounds__(256)
void gemm_wmma(const unsigned int* __restrict__ Ap, const unsigned int* __restrict__ Wp,
               const float* __restrict__ bias, float* __restrict__ Y,
               int M, int nkb, int ldy, int Nc) {
  const int lane    = threadIdx.x & 31;
  const int wave    = threadIdx.x >> 5;
  const int tile_m0 = (blockIdx.y * 8 + wave) * 2;
  if (tile_m0 * 16 >= M) return;            // uniform per-wave exit
  const int tile_m1 = tile_m0 + 1;          // valid: M/16 is always even
  const int n0 = blockIdx.x * 2;
  const int n1 = n0 + 1;

  const uint4* ap0 = (const uint4*)Ap + ((size_t)tile_m0 * nkb * 32 + lane) * 2;
  const uint4* ap1 = (const uint4*)Ap + ((size_t)tile_m1 * nkb * 32 + lane) * 2;
  const uint4* bp0 = (const uint4*)Wp + ((size_t)n0 * nkb * 32 + lane) * 2;
  const uint4* bp1 = (const uint4*)Wp + ((size_t)n1 * nkb * 32 + lane) * 2;

  v8f acc00 = {}, acc01 = {}, acc10 = {}, acc11 = {};
  for (int kbi = 0; kbi < nkb; ++kbi) {
    uint4 a00 = ap0[0], a01 = ap0[1];
    uint4 a10 = ap1[0], a11 = ap1[1];
    uint4 b00 = bp0[0], b01 = bp0[1];
    uint4 b10 = bp1[0], b11 = bp1[1];
    ap0 += 64; ap1 += 64; bp0 += 64; bp1 += 64;   // 256 uints per K-chunk
    v8u au0 = {a00.x, a00.y, a00.z, a00.w, a01.x, a01.y, a01.z, a01.w};
    v8u au1 = {a10.x, a10.y, a10.z, a10.w, a11.x, a11.y, a11.z, a11.w};
    v8u bu0 = {b00.x, b00.y, b00.z, b00.w, b01.x, b01.y, b01.z, b01.w};
    v8u bu1 = {b10.x, b10.y, b10.z, b10.w, b11.x, b11.y, b11.z, b11.w};
    v16bf af0 = __builtin_bit_cast(v16bf, au0);
    v16bf af1 = __builtin_bit_cast(v16bf, au1);
    v16bf bf0 = __builtin_bit_cast(v16bf, bu0);
    v16bf bf1 = __builtin_bit_cast(v16bf, bu1);
    acc00 = __builtin_amdgcn_wmma_f32_16x16x32_bf16(false, af0, false, bf0,
                                                    (short)0, acc00, false, false);
    acc01 = __builtin_amdgcn_wmma_f32_16x16x32_bf16(false, af0, false, bf1,
                                                    (short)0, acc01, false, false);
    acc10 = __builtin_amdgcn_wmma_f32_16x16x32_bf16(false, af1, false, bf0,
                                                    (short)0, acc10, false, false);
    acc11 = __builtin_amdgcn_wmma_f32_16x16x32_bf16(false, af1, false, bf1,
                                                    (short)0, acc11, false, false);
  }

  const int hf   = lane >> 4;
  const int col0 = n0 * 16 + (lane & 15);
  const int col1 = n1 * 16 + (lane & 15);
  const float bv0 = (col0 < Nc) ? bias[col0] : 0.f;
  const float bv1 = (col1 < Nc) ? bias[col1] : 0.f;
#pragma unroll
  for (int r = 0; r < 8; ++r) {
    int m0 = tile_m0 * 16 + r + hf * 8;     // C layout: VGPR r -> M=r (+8 hi half)
    int m1 = tile_m1 * 16 + r + hf * 8;
    Y[(size_t)m0 * ldy + col0] = (col0 < Nc) ? acc00[r] + bv0 : 0.f;
    Y[(size_t)m0 * ldy + col1] = (col1 < Nc) ? acc01[r] + bv1 : 0.f;
    Y[(size_t)m1 * ldy + col0] = (col0 < Nc) ? acc10[r] + bv0 : 0.f;
    Y[(size_t)m1 * ldy + col1] = (col1 < Nc) ? acc11[r] + bv1 : 0.f;
  }
}

// per-channel sum & sumsq over M rows (stride ld) -> stats[c], stats[Nc+c]
__global__ __launch_bounds__(256)
void channel_stats(const float* __restrict__ Y, int M, int Nc, int ld,
                   float* __restrict__ stats) {
  int c = blockIdx.x;
  float s = 0.f, s2 = 0.f;
  for (int m = threadIdx.x; m < M; m += 256) {
    float v = Y[(size_t)m * ld + c];
    s += v; s2 += v * v;
  }
  __shared__ float sh[256], sh2[256];
  sh[threadIdx.x] = s; sh2[threadIdx.x] = s2;
  __syncthreads();
  for (int st = 128; st > 0; st >>= 1) {
    if (threadIdx.x < st) { sh[threadIdx.x] += sh[threadIdx.x + st];
                            sh2[threadIdx.x] += sh2[threadIdx.x + st]; }
    __syncthreads();
  }
  if (threadIdx.x == 0) { stats[c] = sh[0]; stats[Nc + c] = sh2[0]; }
}

__global__ void bn_relu(float* __restrict__ Y, int M, int Nc, int ld,
                        const float* __restrict__ stats,
                        const float* __restrict__ g, const float* __restrict__ e) {
  size_t total = (size_t)M * Nc;
  for (size_t i = blockIdx.x * (size_t)blockDim.x + threadIdx.x; i < total;
       i += (size_t)gridDim.x * blockDim.x) {
    int c = (int)(i % Nc);
    size_t r = i / Nc;
    float mu  = stats[c] / (float)M;
    float var = stats[Nc + c] / (float)M - mu * mu;
    float v = (Y[r * ld + c] - mu) * rsqrtf(var + 1e-5f) * g[c] + e[c];
    Y[r * ld + c] = fmaxf(v, 0.0f);
  }
}

// (B,3,N) -> (B,N,3)
__global__ void transpose_b3n(const float* __restrict__ in, float* __restrict__ out,
                              int n, size_t total) {
  for (size_t i = blockIdx.x * (size_t)blockDim.x + threadIdx.x; i < total;
       i += (size_t)gridDim.x * blockDim.x) {
    int b = (int)(i / ((size_t)n * 3));
    int r = (int)(i % ((size_t)n * 3));
    int p = r / 3, c = r % 3;
    out[i] = in[((size_t)b * 3 + c) * n + p];
  }
}

// farthest point sampling: one block per batch
__global__ __launch_bounds__(256)
void fps_kernel(const float* __restrict__ xyz, int n, int npoint,
                int* __restrict__ out_idx, float* __restrict__ dist) {
  int b = blockIdx.x, t = threadIdx.x;
  __shared__ float sbest[256]; __shared__ int sbidx[256];
  __shared__ float scx, scy, scz; __shared__ int sfar;
  for (int j = t; j < n; j += 256) dist[(size_t)b * n + j] = 1e10f;
  if (t == 0) sfar = 0;
  __syncthreads();
  for (int i = 0; i < npoint; ++i) {
    if (t == 0) {
      int far = sfar;
      out_idx[b * npoint + i] = far;
      scx = xyz[((size_t)b * n + far) * 3 + 0];
      scy = xyz[((size_t)b * n + far) * 3 + 1];
      scz = xyz[((size_t)b * n + far) * 3 + 2];
    }
    __syncthreads();
    float cx = scx, cy = scy, cz = scz;
    float best = -1.f; int bi = 0;
    for (int j = t; j < n; j += 256) {
      float dx = xyz[((size_t)b * n + j) * 3 + 0] - cx;
      float dy = xyz[((size_t)b * n + j) * 3 + 1] - cy;
      float dz = xyz[((size_t)b * n + j) * 3 + 2] - cz;
      float d = dx * dx + dy * dy + dz * dz;
      float dd = fminf(dist[(size_t)b * n + j], d);
      dist[(size_t)b * n + j] = dd;
      if (dd > best) { best = dd; bi = j; }
    }
    sbest[t] = best; sbidx[t] = bi;
    __syncthreads();
    for (int st = 128; st > 0; st >>= 1) {
      if (t < st && sbest[t + st] > sbest[t]) {
        sbest[t] = sbest[t + st]; sbidx[t] = sbidx[t + st];
      }
      __syncthreads();
    }
    if (t == 0) sfar = sbidx[0];
    __syncthreads();
  }
}

__global__ void gather_xyz(const float* __restrict__ xyz, const int* __restrict__ idx,
                           float* __restrict__ out, int n, int S, size_t total) {
  for (size_t i = blockIdx.x * (size_t)blockDim.x + threadIdx.x; i < total;
       i += (size_t)gridDim.x * blockDim.x) {
    int c = (int)(i % 3);
    int s = (int)((i / 3) % S);
    int b = (int)(i / ((size_t)3 * S));
    out[i] = xyz[((size_t)b * n + idx[b * S + s]) * 3 + c];
  }
}

// first-nsample-in-index-order within radius (matches reference sort semantics)
__global__ void ball_query(float r2, int nsample, const float* __restrict__ xyz,
                           const float* __restrict__ new_xyz, int* __restrict__ gidx,
                           int n, int S, int Btot) {
  for (int q = blockIdx.x * blockDim.x + threadIdx.x; q < Btot * S;
       q += gridDim.x * blockDim.x) {
    int b = q / S, s = q % S;
    float qx = new_xyz[((size_t)b * S + s) * 3 + 0];
    float qy = new_xyz[((size_t)b * S + s) * 3 + 1];
    float qz = new_xyz[((size_t)b * S + s) * 3 + 2];
    int* g = gidx + (size_t)q * nsample;
    int cnt = 0;
    for (int j = 0; j < n && cnt < nsample; ++j) {
      float dx = xyz[((size_t)b * n + j) * 3 + 0] - qx;
      float dy = xyz[((size_t)b * n + j) * 3 + 1] - qy;
      float dz = xyz[((size_t)b * n + j) * 3 + 2] - qz;
      if (dx * dx + dy * dy + dz * dz <= r2) g[cnt++] = j;
    }
    int fill = (cnt > 0) ? g[0] : 0;
    for (int k = cnt; k < nsample; ++k) g[k] = fill;
  }
}

// grouped tensor with padded stride: out[b,s,k,c] for c<Ct as reference, else 0
__global__ void group_points(const float* __restrict__ points, const float* __restrict__ xyz,
                             const float* __restrict__ new_xyz, const int* __restrict__ gidx,
                             float* __restrict__ out, int n, int S, int ns, int Cp,
                             int Ctp, size_t total) {
  int Ct = Cp + 3;
  for (size_t i = blockIdx.x * (size_t)blockDim.x + threadIdx.x; i < total;
       i += (size_t)gridDim.x * blockDim.x) {
    int c = (int)(i % Ctp);
    size_t t = i / Ctp;
    int k = (int)(t % ns);
    int s = (int)((t / ns) % S);
    int b = (int)(t / ((size_t)ns * S));
    float v = 0.f;
    if (c < Ct) {
      int j = gidx[((size_t)b * S + s) * ns + k];
      if (c < Cp) v = points[((size_t)b * n + j) * Cp + c];
      else {
        int cc = c - Cp;
        v = xyz[((size_t)b * n + j) * 3 + cc] - new_xyz[((size_t)b * S + s) * 3 + cc];
      }
    }
    out[i] = v;
  }
}

// max over nsample axis (input stride ldin), into strided concat buffer
__global__ void maxpool_ns(const float* __restrict__ in, float* __restrict__ out,
                           int S, int ns, int C, int ldin, int ostride, int ooff,
                           size_t total) {
  for (size_t i = blockIdx.x * (size_t)blockDim.x + threadIdx.x; i < total;
       i += (size_t)gridDim.x * blockDim.x) {
    int c = (int)(i % C);
    int s = (int)((i / C) % S);
    int b = (int)(i / ((size_t)C * S));
    const float* p = in + (((size_t)b * S + s) * ns) * ldin + c;
    float m = p[0];
    for (int k = 1; k < ns; ++k) m = fmaxf(m, p[(size_t)k * ldin]);
    out[((size_t)b * S + s) * ostride + ooff + c] = m;
  }
}

// copy src[rows,Cs] into dst[rows,Ctot] at column offset
__global__ void concat_copy(const float* __restrict__ src, float* __restrict__ dst,
                            int Cs, int Ctot, int coff, size_t total) {
  for (size_t i = blockIdx.x * (size_t)blockDim.x + threadIdx.x; i < total;
       i += (size_t)gridDim.x * blockDim.x) {
    int c = (int)(i % Cs);
    size_t r = i / Cs;
    dst[r * Ctot + coff + c] = src[i];
  }
}

__global__ void zero_cols(float* __restrict__ dst, int ld, int c0, int nc, size_t rows) {
  size_t total = rows * (size_t)nc;
  for (size_t i = blockIdx.x * (size_t)blockDim.x + threadIdx.x; i < total;
       i += (size_t)gridDim.x * blockDim.x) {
    size_t r = i / nc;
    dst[r * ld + c0 + (int)(i % nc)] = 0.f;
  }
}

// fp3 (s==1): broadcast src[b,C] over n1 points into dst columns
__global__ void broadcast_feat(const float* __restrict__ src, float* __restrict__ dst,
                               int n1, int C, int Ctot, int coff, size_t total) {
  for (size_t i = blockIdx.x * (size_t)blockDim.x + threadIdx.x; i < total;
       i += (size_t)gridDim.x * blockDim.x) {
    int c = (int)(i % C);
    int p = (int)((i / C) % n1);
    int b = (int)(i / ((size_t)C * n1));
    dst[((size_t)b * n1 + p) * Ctot + coff + c] = src[(size_t)b * C + c];
  }
}

// 3-NN inverse-distance interpolation
__global__ void knn3_interp(const float* __restrict__ xyz1, const float* __restrict__ xyz2,
                            const float* __restrict__ p2, float* __restrict__ dst,
                            int n1, int s, int C2, int Ctot, int coff, int Btot) {
  for (int q = blockIdx.x * blockDim.x + threadIdx.x; q < Btot * n1;
       q += gridDim.x * blockDim.x) {
    int b = q / n1, i = q % n1;
    float qx = xyz1[((size_t)b * n1 + i) * 3 + 0];
    float qy = xyz1[((size_t)b * n1 + i) * 3 + 1];
    float qz = xyz1[((size_t)b * n1 + i) * 3 + 2];
    float d0 = 1e30f, d1 = 1e30f, d2 = 1e30f;
    int i0 = 0, i1 = 0, i2 = 0;
    for (int j = 0; j < s; ++j) {
      float dx = xyz2[((size_t)b * s + j) * 3 + 0] - qx;
      float dy = xyz2[((size_t)b * s + j) * 3 + 1] - qy;
      float dz = xyz2[((size_t)b * s + j) * 3 + 2] - qz;
      float d = dx * dx + dy * dy + dz * dz;
      if (d < d0)      { d2 = d1; i2 = i1; d1 = d0; i1 = i0; d0 = d; i0 = j; }
      else if (d < d1) { d2 = d1; i2 = i1; d1 = d; i1 = j; }
      else if (d < d2) { d2 = d; i2 = j; }
    }
    float w0 = 1.0f / (d0 + 1e-8f), w1 = 1.0f / (d1 + 1e-8f), w2 = 1.0f / (d2 + 1e-8f);
    float ws = w0 + w1 + w2;
    w0 /= ws; w1 /= ws; w2 /= ws;
    const float* r0 = p2 + ((size_t)b * s + i0) * C2;
    const float* r1 = p2 + ((size_t)b * s + i1) * C2;
    const float* r2 = p2 + ((size_t)b * s + i2) * C2;
    float* o = dst + ((size_t)b * n1 + i) * Ctot + coff;
    for (int c = 0; c < C2; ++c) o[c] = w0 * r0[c] + w1 * r1[c] + w2 * r2[c];
  }
}

__global__ void log_softmax_rows(const float* __restrict__ logits, int ld,
                                 float* __restrict__ out, int rows, int C) {
  for (int r = blockIdx.x * blockDim.x + threadIdx.x; r < rows;
       r += gridDim.x * blockDim.x) {
    const float* p = logits + (size_t)r * ld;
    float mx = p[0];
    for (int c = 1; c < C; ++c) mx = fmaxf(mx, p[c]);
    float s = 0.f;
    for (int c = 0; c < C; ++c) s += expf(p[c] - mx);
    float lse = mx + logf(s);
    for (int c = 0; c < C; ++c) out[(size_t)r * C + c] = p[c] - lse;
  }
}

__global__ void copy_f32(const float* __restrict__ src, float* __restrict__ dst, size_t n) {
  for (size_t i = blockIdx.x * (size_t)blockDim.x + threadIdx.x; i < n;
       i += (size_t)gridDim.x * blockDim.x) dst[i] = src[i];
}

// ---------------------------------------------------------------------------
// Host orchestration
// ---------------------------------------------------------------------------
struct LayerP { const float *w, *b, *g, *e; };

static inline int eb(size_t total) {
  size_t b = (total + 255) / 256;
  return (int)(b > 16384 ? 16384 : (b == 0 ? 1 : b));
}

extern "C" void kernel_launch(void* const* d_in, const int* in_sizes, int n_in,
                              void* d_out, int out_size, void* d_ws, size_t ws_size,
                              hipStream_t stream) {
  (void)in_sizes; (void)n_in; (void)out_size; (void)ws_size;
  const float* xyz_in = (const float*)d_in[0];   // (B,3,N)
  LayerP L[25];
  int pi = 1;
  for (int i = 0; i < 25; ++i) {
    L[i].w = (const float*)d_in[pi++]; L[i].b = (const float*)d_in[pi++];
    L[i].g = (const float*)d_in[pi++]; L[i].e = (const float*)d_in[pi++];
  }
  const float* conv2_w = (const float*)d_in[pi++];
  const float* conv2_b = (const float*)d_in[pi++];

  // workspace bump allocator (all sizes multiples of 4 -> 16B aligned)
  float* wsf = (float*)d_ws;
  size_t off = 0;
  auto alloc = [&](size_t n) { float* p = wsf + off; off += n; return p; };
  float* xyzT  = alloc((size_t)NB * NPTS * 3);
  float* fdist = alloc((size_t)NB * NPTS);
  int*   fps1  = (int*)alloc((size_t)NB * 512);
  float* l1x   = alloc((size_t)NB * 512 * 3);
  int*   fps2  = (int*)alloc((size_t)NB * 128);
  float* l2x   = alloc((size_t)NB * 128 * 3 + 4);
  int*   gidx  = (int*)alloc((size_t)NB * 512 * 128);
  float* l1p   = alloc((size_t)NB * 512 * 320);
  float* l2p   = alloc((size_t)NB * 128 * 512);
  float* l3p   = alloc((size_t)NB * 1024);
  float* fp3o  = alloc((size_t)NB * 128 * 256);
  float* fp2o  = alloc((size_t)NB * 512 * 128);
  float* stats = alloc(2 * 1024);
  unsigned int* Apk = (unsigned int*)alloc((size_t)32 * 1024 * 1024);  // packed A
  unsigned int* Wpk = (unsigned int*)alloc((size_t)1024 * 1024);       // packed W
  float* GA    = alloc((size_t)524288 * 128);
  float* GB    = alloc((size_t)524288 * 128);

  // one conv(+BN+ReLU) layer; X has stride ldx=p32(K), Y gets stride p32(Nc)
  auto mlp_layer = [&](const LayerP& P, const float* X, int ldx, int Kreal,
                       float* Y, int Nc, int M, bool dobn) {
    int ldy = p32(Nc);
    int nkb = ldx / 32, ntn = ldy / 16;
    pack_w<<<eb((size_t)ntn * nkb * 256), 256, 0, stream>>>(P.w, Wpk, Kreal, Nc, nkb, ntn);
    pack_a<<<eb((size_t)(M / 16) * nkb * 32), 256, 0, stream>>>(X, Apk, ldx, nkb, M / 16);
    dim3 g(ldy / 32, (M / 32 + 7) / 8);
    gemm_wmma<<<g, 256, 0, stream>>>(Apk, Wpk, P.b, Y, M, nkb, ldy, Nc);
    if (dobn) {
      channel_stats<<<Nc, 256, 0, stream>>>(Y, M, Nc, ldy, stats);
      bn_relu<<<eb((size_t)M * Nc), 256, 0, stream>>>(Y, M, Nc, ldy, stats, P.g, P.e);
    }
  };
  auto run_mlp = [&](const LayerP* Ls, int nl, const int* ch, int cin, int M,
                     float*& X, float*& Y) {
    int K = cin;
    for (int i = 0; i < nl; ++i) {
      mlp_layer(Ls[i], X, p32(K), K, Y, ch[i], M, true);
      K = ch[i];
      float* t = X; X = Y; Y = t;   // result ends in X (stride p32(last ch))
    }
  };

  // ---- l0: transpose (B,3,N) -> (B,N,3)
  transpose_b3n<<<eb((size_t)NB * NPTS * 3), 256, 0, stream>>>(
      xyz_in, xyzT, NPTS, (size_t)NB * NPTS * 3);

  // ---- SA1 (S=512)
  fps_kernel<<<NB, 256, 0, stream>>>(xyzT, NPTS, 512, fps1, fdist);
  gather_xyz<<<eb((size_t)NB * 512 * 3), 256, 0, stream>>>(
      xyzT, fps1, l1x, NPTS, 512, (size_t)NB * 512 * 3);

  auto sa_scale = [&](const float* pxyz, const float* pts, int n, int Cp,
                      const float* nxyz, int S, float radius, int ns,
                      const LayerP* Ls, const int* ch, float* outbuf,
                      int ostride, int ooff) {
    ball_query<<<eb((size_t)NB * S), 256, 0, stream>>>(
        radius * radius, ns, pxyz, nxyz, gidx, n, S, NB);
    int Ctp = p32(Cp + 3);
    size_t gt = (size_t)NB * S * ns * Ctp;
    group_points<<<eb(gt), 256, 0, stream>>>(pts, pxyz, nxyz, gidx, GA, n, S, ns, Cp,
                                             Ctp, gt);
    float* X = GA; float* Y = GB;
    run_mlp(Ls, 3, ch, Cp + 3, NB * S * ns, X, Y);
    size_t mt = (size_t)NB * S * ch[2];
    maxpool_ns<<<eb(mt), 256, 0, stream>>>(X, outbuf, S, ns, ch[2], p32(ch[2]),
                                           ostride, ooff, mt);
  };

  { int c0[3] = {32, 32, 64};   sa_scale(xyzT, xyzT, NPTS, 3, l1x, 512, 0.1f,  32, &L[0],  c0, l1p, 320, 0);   }
  { int c1[3] = {64, 64, 128};  sa_scale(xyzT, xyzT, NPTS, 3, l1x, 512, 0.2f,  64, &L[3],  c1, l1p, 320, 64);  }
  { int c2[3] = {64, 96, 128};  sa_scale(xyzT, xyzT, NPTS, 3, l1x, 512, 0.4f, 128, &L[6],  c2, l1p, 320, 192); }

  // ---- SA2 (S=128) on (l1x, l1p[320])
  fps_kernel<<<NB, 256, 0, stream>>>(l1x, 512, 128, fps2, fdist);
  gather_xyz<<<eb((size_t)NB * 128 * 3), 256, 0, stream>>>(
      l1x, fps2, l2x, 512, 128, (size_t)NB * 128 * 3);
  { int c3[3] = {128, 128, 256}; sa_scale(l1x, l1p, 512, 320, l2x, 128, 0.4f,  64, &L[9],  c3, l2p, 512, 0);   }
  { int c4[3] = {128, 196, 256}; sa_scale(l1x, l1p, 512, 320, l2x, 128, 0.8f, 128, &L[12], c4, l2p, 512, 256); }

  // ---- SA3: X = concat(l2x[3], l2p[512]) stride 544, MLP 515->256->512->1024, max over 128
  concat_copy<<<eb((size_t)NB * 128 * 3), 256, 0, stream>>>(
      l2x, GA, 3, 544, 0, (size_t)NB * 128 * 3);
  concat_copy<<<eb((size_t)NB * 128 * 512), 256, 0, stream>>>(
      l2p, GA, 512, 544, 3, (size_t)NB * 128 * 512);
  zero_cols<<<eb((size_t)NB * 128 * 29), 256, 0, stream>>>(GA, 544, 515, 29, NB * 128);
  {
    float* X = GA; float* Y = GB;
    int ch[3] = {256, 512, 1024};
    run_mlp(&L[15], 3, ch, 515, NB * 128, X, Y);
    size_t mt = (size_t)NB * 1024;
    maxpool_ns<<<eb(mt), 256, 0, stream>>>(X, l3p, 1, 128, 1024, 1024, 1024, 0, mt);
  }

  // ---- FP3: concat(l2p[512], broadcast l3p[1024]) stride 1536 -> 256 -> 256 (M=1024)
  concat_copy<<<eb((size_t)NB * 128 * 512), 256, 0, stream>>>(
      l2p, GA, 512, 1536, 0, (size_t)NB * 128 * 512);
  broadcast_feat<<<eb((size_t)NB * 128 * 1024), 256, 0, stream>>>(
      l3p, GA, 128, 1024, 1536, 512, (size_t)NB * 128 * 1024);
  {
    float* X = GA; float* Y = GB;
    int ch[2] = {256, 256};
    run_mlp(&L[18], 2, ch, 1536, NB * 128, X, Y);
    copy_f32<<<eb((size_t)NB * 128 * 256), 256, 0, stream>>>(X, fp3o, (size_t)NB * 128 * 256);
  }

  // ---- FP2: concat(l1p[320], knn3 interp[256]) stride 576 -> 256 -> 128 (M=4096)
  concat_copy<<<eb((size_t)NB * 512 * 320), 256, 0, stream>>>(
      l1p, GA, 320, 576, 0, (size_t)NB * 512 * 320);
  knn3_interp<<<eb((size_t)NB * 512), 256, 0, stream>>>(
      l1x, l2x, fp3o, GA, 512, 128, 256, 576, 320, NB);
  {
    float* X = GA; float* Y = GB;
    int ch[2] = {256, 128};
    run_mlp(&L[20], 2, ch, 576, NB * 512, X, Y);
    copy_f32<<<eb((size_t)NB * 512 * 128), 256, 0, stream>>>(X, fp2o, (size_t)NB * 512 * 128);
  }

  // ---- FP1: concat(xyz,xyz, knn3 interp[128]) stride 160 -> 128 -> 128 (M=65536)
  concat_copy<<<eb((size_t)NB * NPTS * 3), 256, 0, stream>>>(
      xyzT, GA, 3, 160, 0, (size_t)NB * NPTS * 3);
  concat_copy<<<eb((size_t)NB * NPTS * 3), 256, 0, stream>>>(
      xyzT, GA, 3, 160, 3, (size_t)NB * NPTS * 3);
  knn3_interp<<<eb((size_t)NB * NPTS), 256, 0, stream>>>(
      xyzT, l1x, fp2o, GA, NPTS, 512, 128, 160, 6, NB);
  zero_cols<<<eb((size_t)NB * NPTS * 26), 256, 0, stream>>>(GA, 160, 134, 26,
                                                            (size_t)NB * NPTS);
  float* X = GA; float* Y = GB;
  {
    int ch[2] = {128, 128};
    run_mlp(&L[22], 2, ch, 134, NB * NPTS, X, Y);  // result back in GA (stride 128)
  }

  // ---- head: conv_bn_relu 128->128 (M=65536), result in GB (stride 128)
  mlp_layer(L[24], X, 128, 128, Y, 128, NB * NPTS, true);

  // ---- conv2: logits = GB @ conv2_w + b (128->50, ld 64), log_softmax -> d_out
  {
    LayerP C2 = { conv2_w, conv2_b, nullptr, nullptr };
    mlp_layer(C2, Y, 128, 128, X, 50, NB * NPTS, false);
  }
  float* out_f = (float*)d_out;
  log_softmax_rows<<<eb((size_t)NB * NPTS), 256, 0, stream>>>(X, 64, out_f, NB * NPTS, 50);

  // ---- tail: l3_points transposed (B,1024,1) == flat (B,1024)
  copy_f32<<<eb((size_t)NB * 1024), 256, 0, stream>>>(
      l3p, out_f + (size_t)NB * NPTS * 50, (size_t)NB * 1024);
}
